// TransformerBlock_45037027066383
// MI455X (gfx1250) — matrix operand order repaired and checked by reading, not compiled, not run on hardware
//
#include <hip/hip_runtime.h>
#include <hip/hip_bf16.h>

typedef __attribute__((ext_vector_type(16))) _Float16 v16h;
typedef __attribute__((ext_vector_type(8)))  float    v8f;
typedef unsigned int uint32x4 __attribute__((ext_vector_type(4)));
typedef int          int32x8  __attribute__((ext_vector_type(8)));
typedef int          int32x4  __attribute__((ext_vector_type(4)));

#if defined(__has_builtin)
#if __has_builtin(__builtin_amdgcn_tensor_load_to_lds) && \
    __has_builtin(__builtin_amdgcn_s_wait_tensorcnt)
#define HAVE_TDM 1
#endif
#endif
#ifndef HAVE_TDM
#define HAVE_TDM 0
#endif

#define B_   16
#define C_   384
#define N_   1024
#define BN_  (B_ * N_)
#define NH_  8
#define HD_  48
#define HID_ 1536
#define QKV_ (3 * C_)

// DPP ROW_XMASK (0x160|mask) 16-lane xor-reductions, pure VALU (no LDS).
#define XMAX(v, m)                                                             \
  v = fmaxf(v, __int_as_float(__builtin_amdgcn_update_dpp(                     \
                   0, __float_as_int(v), 0x160 + (m), 0xf, 0xf, true)))
#define XADD(v, m)                                                             \
  v += __int_as_float(__builtin_amdgcn_update_dpp(                             \
      0, __float_as_int(v), 0x160 + (m), 0xf, 0xf, true))

// ---------------------------------------------------------------------------
// WMMA fragment loaders (CDNA5 16-bit layouts, CDNA5 ISA §7.12.2)
// ---------------------------------------------------------------------------
__device__ __forceinline__ v16h frag_a(const _Float16* p, int stride) {
  const int lane = threadIdx.x & 31;
  const int m  = lane & 15;
  const int kb = (lane >> 4) << 3;          // 0 or 8
  const _Float16* r = p + m * stride + kb;
  v16h o;
#pragma unroll
  for (int i = 0; i < 8; ++i) { o[i] = r[i]; o[8 + i] = r[16 + i]; }
  return o;
}

__device__ __forceinline__ v16h frag_b(const _Float16* p, int stride) {
  const int lane = threadIdx.x & 31;
  const int n  = lane & 15;
  const int kh = (lane >> 4) << 3;          // 0 or 8
  const _Float16* r = p + n * stride + kh;
  v16h o;
#pragma unroll
  for (int i = 0; i < 8; ++i) { o[i] = r[i]; o[8 + i] = r[16 + i]; }
  return o;
}

// ---------------------------------------------------------------------------
// Generic WMMA GEMM: Y[M,Nout] = A[M,K] * W[Nout,K]^T + bias, fused epilogue.
// A tile: register double-buffered global->LDS staging.
// W tile: Tensor Data Mover DMA into double-buffered LDS (TENSORcnt pipelined),
// falling back to VGPR staging when the TDM builtin is unavailable.
// ---------------------------------------------------------------------------
template <bool GELU, bool F16OUT, bool FADD>
__global__ __launch_bounds__(256)
void gemm_wmma(const _Float16* __restrict__ A, const _Float16* __restrict__ W,
               const float* __restrict__ bias, _Float16* __restrict__ out16,
               float* __restrict__ outf, int M, int Nout, int K) {
  __shared__ __attribute__((aligned(16)))  _Float16 As[128 * 32];
#if HAVE_TDM
  __shared__ __attribute__((aligned(128))) _Float16 Bs[2][64 * 32];
#else
  __shared__ __attribute__((aligned(16)))  _Float16 Bs[1][64 * 32];
#endif
  const int m0 = blockIdx.y * 128;
  const int n0 = blockIdx.x * 64;
  const int t = threadIdx.x;
  const int w = t >> 5, lane = t & 31;

  const int ar0 = t >> 2,          ao0 = (t & 3) << 3;          // slot t
  const int ar1 = (t + 256) >> 2,  ao1 = ((t + 256) & 3) << 3;  // slot t+256

  auto ldA = [&](int k0, int row, int off) {
    return *(const uint4*)&A[(size_t)(m0 + row) * K + k0 + off];
  };

#if HAVE_TDM
  // TDM D# (CDNA5 ISA): 2D tile, 64 rows x 32 halves,
  // data_size=2B, row stride = K elements.  group0: count|lds|global|type=2.
  // 6-arg builtin form (clang-23 lane): (g0, g1, g2, g3, g4, cpol).
  auto tdm_issue = [&](int k0, int buf) {
    const unsigned lds = (unsigned)(uintptr_t)&Bs[buf][0];
    const unsigned long long ga =
        (unsigned long long)(uintptr_t)&W[(size_t)n0 * K + k0];
    uint32x4 g0;
    g0[0] = 1u;                                  // count=1 (valid descriptor)
    g0[1] = lds;                                 // lds_addr
    g0[2] = (unsigned)ga;                        // global_addr[31:0]
    g0[3] = (unsigned)(ga >> 32) | (2u << 30);   // global_addr[56:32] | type=2
    int32x8 g1;
    g1[0] = 0x10000;                             // data_size=1 (2 bytes)
    g1[1] = (K & 0xffff) << 16;                  // tensor_dim0[15:0]
    g1[2] = ((K >> 16) & 0xffff) | ((Nout & 0xffff) << 16);  // dim0 hi|dim1 lo
    g1[3] = ((Nout >> 16) & 0xffff) | (32 << 16);            // dim1 hi|tile0=32
    g1[4] = 64;                                  // tile_dim1=64 (tile_dim2=0)
    g1[5] = K;                                   // tensor_dim0_stride[31:0]
    g1[6] = 0;
    g1[7] = 0;
    int32x4 z; z[0] = 0; z[1] = 0; z[2] = 0; z[3] = 0;
    int32x8 z8;
#pragma unroll
    for (int i = 0; i < 8; ++i) z8[i] = 0;
    __builtin_amdgcn_tensor_load_to_lds(g0, g1, z, z, z8, 0);
  };
  if (w == 0) tdm_issue(0, 0);
#else
  const int br = t >> 2, bo = (t & 3) << 3;
  auto ldB = [&](int k0) {
    return *(const uint4*)&W[(size_t)(n0 + br) * K + k0 + bo];
  };
  uint4 rb = ldB(0);
#endif

  uint4 ra0 = ldA(0, ar0, ao0), ra1 = ldA(0, ar1, ao1);

  v8f acc[4];
#pragma unroll
  for (int j = 0; j < 4; ++j)
#pragma unroll
    for (int r = 0; r < 8; ++r) acc[j][r] = 0.f;

  for (int k0 = 0, it = 0; k0 < K; k0 += 32, ++it) {
    __syncthreads();
    *(uint4*)&As[ar0 * 32 + ao0] = ra0;
    *(uint4*)&As[ar1 * 32 + ao1] = ra1;
#if HAVE_TDM
    if (w == 0) {
      if (k0 + 32 < K) {                 // keep one DMA in flight
        tdm_issue(k0 + 32, (it + 1) & 1);
        __builtin_amdgcn_s_wait_tensorcnt(1);
      } else {
        __builtin_amdgcn_s_wait_tensorcnt(0);
      }
    }
#else
    *(uint4*)&Bs[0][br * 32 + bo] = rb;
#endif
    __syncthreads();
    if (k0 + 32 < K) {                   // overlap next A tile with WMMA
      ra0 = ldA(k0 + 32, ar0, ao0);
      ra1 = ldA(k0 + 32, ar1, ao1);
#if !HAVE_TDM
      rb = ldB(k0 + 32);
#endif
      if (k0 + 64 < K)
        __builtin_prefetch(&A[(size_t)(m0 + ar0) * K + k0 + 64], 0, 1);
    }
#if HAVE_TDM
    const _Float16* bb = Bs[it & 1];
#else
    const _Float16* bb = Bs[0];
#endif
    const v16h a  = frag_a(As + w * 16 * 32, 32);
    const v16h b0 = frag_b(bb,               32);
    const v16h b1 = frag_b(bb + 1 * 16 * 32, 32);
    const v16h b2 = frag_b(bb + 2 * 16 * 32, 32);
    const v16h b3 = frag_b(bb + 3 * 16 * 32, 32);
    acc[0] = __builtin_amdgcn_wmma_f32_16x16x32_f16(false, a, false, b0, (short)0, acc[0], false, false);
    acc[1] = __builtin_amdgcn_wmma_f32_16x16x32_f16(false, a, false, b1, (short)0, acc[1], false, false);
    acc[2] = __builtin_amdgcn_wmma_f32_16x16x32_f16(false, a, false, b2, (short)0, acc[2], false, false);
    acc[3] = __builtin_amdgcn_wmma_f32_16x16x32_f16(false, a, false, b3, (short)0, acc[3], false, false);
  }

  const int grp = (lane >> 4) << 3;
#pragma unroll
  for (int j = 0; j < 4; ++j) {
    const int col = n0 + j * 16 + (lane & 15);
    const float bv = bias[col];
#pragma unroll
    for (int r = 0; r < 8; ++r) {
      const int row = m0 + w * 16 + r + grp;
      float v = acc[j][r] + bv;
      if (GELU) v = 0.5f * v * (1.f + erff(v * 0.70710678f));
      if (F16OUT) out16[(size_t)row * Nout + col] = (_Float16)v;
      if (FADD)  outf[(size_t)row * Nout + col] += v;
    }
  }
}

// ---------------------------------------------------------------------------
// Flash attention, one (batch, head, 128-query-block) per 256-thread block.
// ---------------------------------------------------------------------------
__global__ __launch_bounds__(256)
void attn_wmma(const _Float16* __restrict__ qkv, _Float16* __restrict__ attn) {
  __shared__ __attribute__((aligned(16))) _Float16 Qs[128 * 64];
  __shared__ __attribute__((aligned(16))) _Float16 Ks[32 * 64];
  __shared__ __attribute__((aligned(16))) _Float16 Vs[32 * 64];
  __shared__ __attribute__((aligned(16))) _Float16 Vts[48 * 32];
  __shared__ __attribute__((aligned(16))) _Float16 Ps[8][16 * 32];

  const int qb = blockIdx.x & 7;
  const int bh = blockIdx.x >> 3;
  const int h = bh & 7, b = bh >> 3;
  const int t0 = b * N_ + qb * 128;
  const int t = threadIdx.x, w = t >> 5, lane = t & 31;
  const uint4 z4 = make_uint4(0u, 0u, 0u, 0u);

#pragma unroll
  for (int i = 0; i < 4; ++i) {                   // stage Q (pad d>=48 -> 0)
    int slot = t + 256 * i;
    int row = slot >> 3, off = (slot & 7) << 3;
    uint4 v = z4;
    if (off < HD_)
      v = *(const uint4*)&qkv[(size_t)(t0 + row) * QKV_ + h * HD_ + off];
    *(uint4*)&Qs[row * 64 + off] = v;
  }
  __syncthreads();
  const v16h qa0 = frag_a(Qs + w * 16 * 64, 64);
  const v16h qa1 = frag_a(Qs + w * 16 * 64 + 32, 64);

  v8f o0, o1, o2;
  float mreg[8], lreg[8];
#pragma unroll
  for (int r = 0; r < 8; ++r) {
    o0[r] = 0.f; o1[r] = 0.f; o2[r] = 0.f;
    mreg[r] = -3.4e38f; lreg[r] = 0.f;
  }
  const float scale = 0.1443375673f;              // 1/sqrt(48)
  const int key = t >> 3, koff = (t & 7) << 3;    // staging coords (32 x 64)

  for (int kb = 0; kb < N_ / 32; ++kb) {
    const int key0 = kb * 32;
    __syncthreads();
    {                                             // stage K & V, coalesced
      const size_t rowb = (size_t)(b * N_ + key0 + key) * QKV_ + h * HD_;
      uint4 vk = z4, vv = z4;
      if (koff < HD_) {
        vk = *(const uint4*)&qkv[rowb + C_ + koff];
        vv = *(const uint4*)&qkv[rowb + 2 * C_ + koff];
      }
      *(uint4*)&Ks[key * 64 + koff] = vk;
      *(uint4*)&Vs[key * 64 + koff] = vv;
    }
    if (kb + 1 < N_ / 32) {                       // prefetch next key block
      const size_t nrow = (size_t)(b * N_ + key0 + 32 + key) * QKV_ + h * HD_;
      __builtin_prefetch(&qkv[nrow + C_], 0, 1);
      __builtin_prefetch(&qkv[nrow + 2 * C_], 0, 1);
    }
    __syncthreads();
#pragma unroll
    for (int i = 0; i < 6; ++i) {                 // transpose V -> Vts[48][32]
      int idx = t + 256 * i;
      int d = idx >> 5, kk = idx & 31;
      Vts[d * 32 + kk] = Vs[kk * 64 + d];
    }

    v8f s0, s1;
#pragma unroll
    for (int r = 0; r < 8; ++r) { s0[r] = 0.f; s1[r] = 0.f; }
    {                                             // preload 4 frags, 4 WMMAs
      const v16h bk0 = frag_b(Ks, 64);
      const v16h bk1 = frag_b(Ks + 32, 64);
      const v16h bk2 = frag_b(Ks + 16 * 64, 64);
      const v16h bk3 = frag_b(Ks + 16 * 64 + 32, 64);
      s0 = __builtin_amdgcn_wmma_f32_16x16x32_f16(false, qa0, false, bk0, (short)0, s0, false, false);
      s0 = __builtin_amdgcn_wmma_f32_16x16x32_f16(false, qa1, false, bk1, (short)0, s0, false, false);
      s1 = __builtin_amdgcn_wmma_f32_16x16x32_f16(false, qa0, false, bk2, (short)0, s1, false, false);
      s1 = __builtin_amdgcn_wmma_f32_16x16x32_f16(false, qa1, false, bk3, (short)0, s1, false, false);
    }

#pragma unroll
    for (int r = 0; r < 8; ++r) {                 // online softmax per row
      float x0 = s0[r] * scale, x1 = s1[r] * scale;
      float mb = fmaxf(x0, x1);
      XMAX(mb, 1); XMAX(mb, 2); XMAX(mb, 4); XMAX(mb, 8);
      const float mnew = fmaxf(mreg[r], mb);
      const float alpha = __expf(mreg[r] - mnew);
      mreg[r] = mnew;
      const float p0 = __expf(x0 - mnew), p1 = __expf(x1 - mnew);
      float rs = p0 + p1;
      XADD(rs, 1); XADD(rs, 2); XADD(rs, 4); XADD(rs, 8);
      lreg[r] = lreg[r] * alpha + rs;
      const int row = r + ((lane >> 4) << 3);
      Ps[w][row * 32 + (lane & 15)]      = (_Float16)p0;
      Ps[w][row * 32 + 16 + (lane & 15)] = (_Float16)p1;
      o0[r] *= alpha; o1[r] *= alpha; o2[r] *= alpha;
    }
    __syncthreads();                              // Vts visible to all waves

    const v16h pa  = frag_a(Ps[w], 32);
    const v16h vb0 = frag_b(Vts, 32);
    const v16h vb1 = frag_b(Vts + 16 * 32, 32);
    const v16h vb2 = frag_b(Vts + 32 * 32, 32);
    o0 = __builtin_amdgcn_wmma_f32_16x16x32_f16(false, pa, false, vb0, (short)0, o0, false, false);
    o1 = __builtin_amdgcn_wmma_f32_16x16x32_f16(false, pa, false, vb1, (short)0, o1, false, false);
    o2 = __builtin_amdgcn_wmma_f32_16x16x32_f16(false, pa, false, vb2, (short)0, o2, false, false);
  }

#pragma unroll
  for (int r = 0; r < 8; ++r) {
    const int row = t0 + w * 16 + r + ((lane >> 4) << 3);
    const float inv = 1.f / lreg[r];
    attn[(size_t)row * C_ + h * HD_ + (lane & 15)]      = (_Float16)(o0[r] * inv);
    attn[(size_t)row * C_ + h * HD_ + 16 + (lane & 15)] = (_Float16)(o1[r] * inv);
    attn[(size_t)row * C_ + h * HD_ + 32 + (lane & 15)] = (_Float16)(o2[r] * inv);
  }
}

// ---------------------------------------------------------------------------
// LayerNorm (one token per 128-thread block). FIRST=true also does the
// (B,C,N)->(B,N,C) gather and emits the fp32 residual token buffer.
// ---------------------------------------------------------------------------
template <bool FIRST>
__global__ __launch_bounds__(128)
void ln_kernel(const float* __restrict__ x, const float* __restrict__ wgt,
               const float* __restrict__ bia, _Float16* __restrict__ out,
               float* __restrict__ xtok) {
  const int tok = blockIdx.x;
  const int bb = tok >> 10, n = tok & 1023;
  __shared__ float red[2][4];
  float vals[3], s = 0.f, ss = 0.f;
#pragma unroll
  for (int i = 0; i < 3; ++i) {
    const int c = threadIdx.x + 128 * i;
    const float v = FIRST ? x[((size_t)bb * C_ + c) * N_ + n]
                          : x[(size_t)tok * C_ + c];
    vals[i] = v; s += v; ss += v * v;
  }
#pragma unroll
  for (int m = 16; m; m >>= 1) { s += __shfl_xor(s, m); ss += __shfl_xor(ss, m); }
  if ((threadIdx.x & 31) == 0) {
    red[0][threadIdx.x >> 5] = s; red[1][threadIdx.x >> 5] = ss;
  }
  __syncthreads();
  s  = red[0][0] + red[0][1] + red[0][2] + red[0][3];
  ss = red[1][0] + red[1][1] + red[1][2] + red[1][3];
  const float mu = s * (1.f / C_);
  const float rstd = rsqrtf(ss * (1.f / C_) - mu * mu + 1e-5f);
#pragma unroll
  for (int i = 0; i < 3; ++i) {
    const int c = threadIdx.x + 128 * i;
    out[(size_t)tok * C_ + c] = (_Float16)((vals[i] - mu) * rstd * wgt[c] + bia[c]);
    if (FIRST) xtok[(size_t)tok * C_ + c] = vals[i];
  }
}

__global__ __launch_bounds__(256)
void cvt_f16(const float* __restrict__ a, _Float16* __restrict__ o, int n) {
  const int i = blockIdx.x * 256 + threadIdx.x;
  if (i < n) o[i] = (_Float16)a[i];
}

__global__ __launch_bounds__(256)
void transpose_out(const float* __restrict__ xt, float* __restrict__ o) {
  const int i = blockIdx.x * 256 + threadIdx.x;     // (B,C,N) linear
  const int b = i / (C_ * N_);
  const int rem = i - b * (C_ * N_);
  const int c = rem / N_;
  const int n = rem - c * N_;
  o[i] = xt[((size_t)b * N_ + n) * C_ + c];
}

// ---------------------------------------------------------------------------
extern "C" void kernel_launch(void* const* d_in, const int* in_sizes, int n_in,
                              void* d_out, int out_size, void* d_ws,
                              size_t ws_size, hipStream_t stream) {
  (void)in_sizes; (void)n_in; (void)out_size; (void)ws_size;
  const float* x     = (const float*)d_in[0];
  const float* ln1w  = (const float*)d_in[1];
  const float* ln1b  = (const float*)d_in[2];
  const float* qkvw  = (const float*)d_in[3];
  const float* qkvb  = (const float*)d_in[4];
  const float* projw = (const float*)d_in[5];
  const float* projb = (const float*)d_in[6];
  const float* ln2w  = (const float*)d_in[7];
  const float* ln2b  = (const float*)d_in[8];
  const float* fc1w  = (const float*)d_in[9];
  const float* fc1b  = (const float*)d_in[10];
  const float* fc2w  = (const float*)d_in[11];
  const float* fc2b  = (const float*)d_in[12];
  float* out = (float*)d_out;

  char* p = (char*)d_ws;
  auto alloc = [&](size_t bytes) -> char* {
    char* r = p; p += (bytes + 255) & ~(size_t)255; return r;
  };
  float*    xtok  = (float*)   alloc((size_t)BN_ * C_ * 4);   // fp32 residual
  _Float16* hbuf  = (_Float16*)alloc((size_t)BN_ * C_ * 2);   // LN out (reused)
  _Float16* qkvh  = (_Float16*)alloc((size_t)BN_ * QKV_ * 2);
  _Float16* attnh = (_Float16*)alloc((size_t)BN_ * C_ * 2);
  _Float16* hidh  = (_Float16*)alloc((size_t)BN_ * HID_ * 2);
  _Float16* wq    = (_Float16*)alloc((size_t)QKV_ * C_ * 2);
  _Float16* wp    = (_Float16*)alloc((size_t)C_ * C_ * 2);
  _Float16* w1    = (_Float16*)alloc((size_t)HID_ * C_ * 2);
  _Float16* w2    = (_Float16*)alloc((size_t)C_ * HID_ * 2);

  cvt_f16<<<(QKV_ * C_ + 255) / 256, 256, 0, stream>>>(qkvw, wq, QKV_ * C_);
  cvt_f16<<<(C_ * C_ + 255) / 256, 256, 0, stream>>>(projw, wp, C_ * C_);
  cvt_f16<<<(HID_ * C_ + 255) / 256, 256, 0, stream>>>(fc1w, w1, HID_ * C_);
  cvt_f16<<<(C_ * HID_ + 255) / 256, 256, 0, stream>>>(fc2w, w2, C_ * HID_);

  ln_kernel<true><<<BN_, 128, 0, stream>>>(x, ln1w, ln1b, hbuf, xtok);

  gemm_wmma<false, true, false><<<dim3(QKV_ / 64, BN_ / 128), 256, 0, stream>>>(
      hbuf, wq, qkvb, qkvh, nullptr, BN_, QKV_, C_);

  attn_wmma<<<B_ * NH_ * (N_ / 128), 256, 0, stream>>>(qkvh, attnh);

  gemm_wmma<false, false, true><<<dim3(C_ / 64, BN_ / 128), 256, 0, stream>>>(
      attnh, wp, projb, nullptr, xtok, BN_, C_, C_);

  ln_kernel<false><<<BN_, 128, 0, stream>>>(xtok, ln2w, ln2b, hbuf, nullptr);

  gemm_wmma<true, true, false><<<dim3(HID_ / 64, BN_ / 128), 256, 0, stream>>>(
      hbuf, w1, fc1b, hidh, nullptr, BN_, HID_, C_);

  gemm_wmma<false, false, true><<<dim3(C_ / 64, BN_ / 128), 256, 0, stream>>>(
      hidh, w2, fc2b, nullptr, xtok, BN_, C_, HID_);

  transpose_out<<<(BN_ * C_) / 256, 256, 0, stream>>>(xtok, out);
}
